// KMultiheadAttention_13554916786206
// MI455X (gfx1250) — matrix operand order repaired and checked by reading, not compiled, not run on hardware
//
#include <hip/hip_runtime.h>

typedef __attribute__((ext_vector_type(16))) _Float16 v16h;
typedef __attribute__((ext_vector_type(8)))  float    v8f;
typedef __attribute__((ext_vector_type(2)))  _Float16 h2;
typedef int v4i __attribute__((vector_size(16)));

#define E_DIM  1024
#define S_DIM  2048
#define B_DIM  2
#define H_DIM  16
#define DK_DIM 64
#define M_TOTAL (B_DIM * S_DIM)   // 4096

#ifndef __has_builtin
#define __has_builtin(x) 0
#endif
#if __has_builtin(__builtin_amdgcn_global_load_async_to_lds_b128) && \
    __has_builtin(__builtin_amdgcn_s_wait_asynccnt)
#define ASYNC_LDS 1
#else
#define ASYNC_LDS 0
#endif

// 16-byte global -> LDS copy. Uses the CDNA5 async-LDS path (ASYNCcnt,
// no VGPR round trip) when the builtin exists; else a plain b128 round trip.
__device__ __forceinline__ void cp16(void* lds, const void* g) {
#if ASYNC_LDS
  __builtin_amdgcn_global_load_async_to_lds_b128(
      (__attribute__((address_space(1))) v4i*)const_cast<void*>(g),
      (__attribute__((address_space(3))) v4i*)lds, 0, 0);
#else
  *(int4*)lds = *(const int4*)g;
#endif
}
__device__ __forceinline__ void cp16_wait() {
#if ASYNC_LDS
  __builtin_amdgcn_s_wait_asynccnt(0);
#endif
}

// ---------------------------------------------------------------------------
// Fragment loaders matching the CDNA5 wave32 WMMA VGPR layouts (ISA 7.12.2).
// A (16x32 f16): lane holds row M=lane&15; sel=lane>>4.
//   halves 0..7  -> K = k0 + sel*8 + 0..7 ; halves 8..15 -> +16
// B (32x16 f16): lane holds col N=lane&15; same K order as A.
// ---------------------------------------------------------------------------
__device__ __forceinline__ v16h ld_frag_a(const _Float16* base, int ld,
                                          int row0, int k0, int lane) {
  const int r   = row0 + (lane & 15);
  const int sel = (lane >> 4) & 1;
  const _Float16* p = base + r * ld + k0 + sel * 8;
  v16h a;
#pragma unroll
  for (int q = 0; q < 4; ++q) {
    h2 x = *(const h2*)(p + 2 * q);
    h2 y = *(const h2*)(p + 16 + 2 * q);
    a[2 * q]     = x[0];
    a[2 * q + 1] = x[1];
    a[8 + 2 * q] = y[0];
    a[9 + 2 * q] = y[1];
  }
  return a;
}

__device__ __forceinline__ v16h ld_frag_b(const _Float16* base, int ld,
                                          int n0, int k0, int lane) {
  const int n   = n0 + (lane & 15);
  const int sel = (lane >> 4) & 1;
  const _Float16* p = base + n * ld + k0 + sel * 8;
  v16h b;
#pragma unroll
  for (int q = 0; q < 8; ++q) {
    h2 x = *(const h2*)(p + 2 * (q & 3) + (q >> 2) * 16);
    b[2 * q]     = x[0];
    b[2 * q + 1] = x[1];
  }
  return b;
}

__device__ __forceinline__ v8f wmma_f16(v16h a, v16h b, v8f c) {
  return __builtin_amdgcn_wmma_f32_16x16x32_f16(false, a, false, b,
                                                (short)0, c, false, false);
}

// ---------------------------------------------------------------------------
// Kernel 0: one-shot weight convert + transpose. W f32 [K][N] -> WT f16 [N][K].
// Makes every GEMM B-tile a contiguous f16 copy (async-LDS eligible) and
// removes per-block f32->f16 conversion from the GEMM hot loops.
// ---------------------------------------------------------------------------
__global__ __launch_bounds__(256) void k_wcvt(const float* __restrict__ W,
                                              _Float16* __restrict__ WT) {
  __shared__ _Float16 tl[32][34];
  const int t  = threadIdx.x;
  const int n0 = blockIdx.x * 32;
  const int k0 = blockIdx.y * 32;
  const int tn = t & 31;
  const int tk = t >> 5;  // 0..7
#pragma unroll
  for (int p = 0; p < 4; ++p) {
    const int k = tk + p * 8;
    tl[tn][k] = (_Float16)W[(size_t)(k0 + k) * E_DIM + n0 + tn];
  }
  __syncthreads();
  const int wn = t >> 3;        // 0..31
  const int wc = (t & 7) * 4;   // 4 halves each
#pragma unroll
  for (int i = 0; i < 4; ++i)
    WT[(size_t)(n0 + wn) * E_DIM + k0 + wc + i] = tl[wn][wc + i];
}

// ---------------------------------------------------------------------------
// Kernel 1: projection GEMM. Y = (X * W + bias) * scale, written head-split
// f16 [B,H,S,DK]. X:[4096,1024] f32 (cvt on load), WT:[N][K] f16 (async copy).
// Block 256 thr (8 waves), tile 128x128, k-step 32, wave tile 32x64.
// ---------------------------------------------------------------------------
__global__ __launch_bounds__(256) void k_proj_gemm(
    const float* __restrict__ X, const _Float16* __restrict__ WT,
    const float* __restrict__ bias, _Float16* __restrict__ Y, float scale) {
  __shared__ alignas(16) _Float16 As[128][40];   // [m][k]
  __shared__ alignas(16) _Float16 Bs[128][40];   // [n][k]

  const int t    = threadIdx.x;
  const int lane = t & 31;
  const int wid  = t >> 5;
  const int wm   = wid >> 1;
  const int wn   = wid & 1;
  const int m0   = blockIdx.x * 128;
  const int n0   = blockIdx.y * 128;
  const int sel  = (lane >> 4) & 1;

  v8f acc[2][4];
  v8f z = {};
#pragma unroll
  for (int i = 0; i < 2; ++i)
#pragma unroll
    for (int j = 0; j < 4; ++j) acc[i][j] = z;

  const int arow = t >> 1, ac0 = (t & 1) * 16;  // A: 2 thr/row, 16 f32 each
  const int brow = t >> 1, bhr = (t & 1) * 16;  // B: 2 thr/row, 16 f16 each

  for (int k0 = 0; k0 < E_DIM; k0 += 32) {
    __syncthreads();
    // A tile: f32 -> f16 convert
    const float* xa = X + (size_t)(m0 + arow) * E_DIM + k0 + ac0;
#pragma unroll
    for (int i = 0; i < 16; ++i) As[arow][ac0 + i] = (_Float16)xa[i];
    // B tile: straight f16 copy (async to LDS on CDNA5)
    const _Float16* xw = WT + (size_t)(n0 + brow) * E_DIM + k0 + bhr;
#pragma unroll
    for (int c = 0; c < 2; ++c) cp16(&Bs[brow][bhr + c * 8], xw + c * 8);
    if (k0 + 32 < E_DIM) __builtin_prefetch(xa + 32, 0, 1);
    cp16_wait();
    __syncthreads();

    v16h af[2];
#pragma unroll
    for (int mt = 0; mt < 2; ++mt)
      af[mt] = ld_frag_a(&As[0][0], 40, wm * 32 + mt * 16, 0, lane);
#pragma unroll
    for (int nt = 0; nt < 4; ++nt) {
      v16h bf = ld_frag_b(&Bs[0][0], 40, wn * 64 + nt * 16, 0, lane);
#pragma unroll
      for (int mt = 0; mt < 2; ++mt)
        acc[mt][nt] = wmma_f16(af[mt], bf, acc[mt][nt]);
    }
  }

#pragma unroll
  for (int mt = 0; mt < 2; ++mt)
#pragma unroll
    for (int nt = 0; nt < 4; ++nt) {
      const int col = n0 + wn * 64 + nt * 16 + (lane & 15);
      const float bb = bias[col];
      const int h = col >> 6, dk = col & 63;
#pragma unroll
      for (int j = 0; j < 8; ++j) {
        const int row = m0 + wm * 32 + mt * 16 + j + sel * 8;
        const int b = row >> 11, s = row & (S_DIM - 1);
        const size_t idx =
            (((size_t)(b * H_DIM + h)) * S_DIM + s) * DK_DIM + dk;
        Y[idx] = (_Float16)((acc[mt][nt][j] + bb) * scale);
      }
    }
}

// ---------------------------------------------------------------------------
// Kernel 2: flash attention, causal. One block = 128 q-rows of one (b,h).
// Q/K tiles staged via async-LDS; V transposed through VGPRs.
// ---------------------------------------------------------------------------
__global__ __launch_bounds__(128) void k_attn(
    const _Float16* __restrict__ Qp, const _Float16* __restrict__ Kp,
    const _Float16* __restrict__ Vp, _Float16* __restrict__ Ctx) {
  __shared__ alignas(16) _Float16 Qs[128][72];   // [q][d]
  __shared__ alignas(16) _Float16 Ks[64][72];    // [key][d]
  __shared__ alignas(16) _Float16 Vs[64][72];    // [dk][key]
  __shared__ alignas(16) _Float16 Ps[4][16][72]; // per-wave P staging

  const int t    = threadIdx.x;
  const int lane = t & 31;
  const int wid  = t >> 5;
  const int sel  = (lane >> 4) & 1;
  const int q0   = blockIdx.x * 128;
  const int bh   = blockIdx.y;
  const int b    = bh >> 4, h = bh & 15;
  const int wrow0 = wid * 32;

  { // stage Q tile once: 128 rows x 128B, 8 async b128 per thread
    const _Float16* src = Qp + ((size_t)bh * S_DIM + q0 + t) * DK_DIM;
#pragma unroll
    for (int i = 0; i < 8; ++i) cp16(&Qs[t][i * 8], src + i * 8);
  }

  v8f ctx[2][4];
  v8f z = {};
#pragma unroll
  for (int mt = 0; mt < 2; ++mt)
#pragma unroll
    for (int nt = 0; nt < 4; ++nt) ctx[mt][nt] = z;
  float mst[2][8], lst[2][8];
#pragma unroll
  for (int mt = 0; mt < 2; ++mt)
#pragma unroll
    for (int j = 0; j < 8; ++j) { mst[mt][j] = -3.0e38f; lst[mt][j] = 0.0f; }

  const int nkt = (q0 >> 6) + 2;   // causal: tiles with k0 <= q0+127

  for (int kt = 0; kt < nkt; ++kt) {
    const int k0 = kt * 64;
    __syncthreads();
    { // K tile: async b128; V tile: VGPR transpose [key][dk] -> [dk][key]
      const int r = t >> 1, c0 = (t & 1) * 32;
      const _Float16* sk = Kp + ((size_t)bh * S_DIM + k0 + r) * DK_DIM + c0;
#pragma unroll
      for (int c = 0; c < 4; ++c) cp16(&Ks[r][c0 + c * 8], sk + c * 8);
      const _Float16* sv = Vp + ((size_t)bh * S_DIM + k0 + r) * DK_DIM + c0;
#pragma unroll
      for (int i = 0; i < 32; ++i) Vs[c0 + i][r] = sv[i];
    }
    cp16_wait();
    __syncthreads();

#pragma unroll
    for (int mt = 0; mt < 2; ++mt) {
      // S = Q * K^T  (K-dim 64 -> two k=32 WMMA steps)
      v8f sc[4];
#pragma unroll
      for (int nt = 0; nt < 4; ++nt) sc[nt] = z;
#pragma unroll
      for (int kc = 0; kc < 2; ++kc) {
        v16h aq = ld_frag_a(&Qs[0][0], 72, wrow0 + mt * 16, kc * 32, lane);
#pragma unroll
        for (int nt = 0; nt < 4; ++nt) {
          v16h bk = ld_frag_b(&Ks[0][0], 72, nt * 16, kc * 32, lane);
          sc[nt] = wmma_f16(aq, bk, sc[nt]);
        }
      }
      // causal mask
#pragma unroll
      for (int nt = 0; nt < 4; ++nt) {
        const int col = k0 + nt * 16 + (lane & 15);
#pragma unroll
        for (int j = 0; j < 8; ++j) {
          const int row = q0 + wrow0 + mt * 16 + j + sel * 8;
          if (col > row) sc[nt][j] = -1.0e30f;
        }
      }
      // online softmax (row reductions stay inside each 16-lane half)
      float corr[8], snew[8];
#pragma unroll
      for (int j = 0; j < 8; ++j) {
        float v = sc[0][j];
#pragma unroll
        for (int nt = 1; nt < 4; ++nt) v = fmaxf(v, sc[nt][j]);
        v = fmaxf(v, __shfl_xor(v, 8, 32));
        v = fmaxf(v, __shfl_xor(v, 4, 32));
        v = fmaxf(v, __shfl_xor(v, 2, 32));
        v = fmaxf(v, __shfl_xor(v, 1, 32));
        const float mnew = fmaxf(mst[mt][j], v);
        corr[j] = __expf(mst[mt][j] - mnew);
        mst[mt][j] = mnew;
        snew[j] = 0.0f;
      }
      // P = exp(S - m); C-layout -> A-layout via per-wave LDS
#pragma unroll
      for (int nt = 0; nt < 4; ++nt) {
        const int c = nt * 16 + (lane & 15);
#pragma unroll
        for (int j = 0; j < 8; ++j) {
          const float p = __expf(sc[nt][j] - mst[mt][j]);
          snew[j] += p;
          Ps[wid][j + sel * 8][c] = (_Float16)p;
        }
      }
#pragma unroll
      for (int j = 0; j < 8; ++j) {
        float s = snew[j];
        s += __shfl_xor(s, 8, 32);
        s += __shfl_xor(s, 4, 32);
        s += __shfl_xor(s, 2, 32);
        s += __shfl_xor(s, 1, 32);
        lst[mt][j] = lst[mt][j] * corr[j] + s;
      }
#pragma unroll
      for (int nt = 0; nt < 4; ++nt)
#pragma unroll
        for (int j = 0; j < 8; ++j) ctx[mt][nt][j] *= corr[j];
      // ctx += P * V
#pragma unroll
      for (int kc = 0; kc < 2; ++kc) {
        v16h ap = ld_frag_a(&Ps[wid][0][0], 72, 0, kc * 32, lane);
#pragma unroll
        for (int nt = 0; nt < 4; ++nt) {
          v16h bv = ld_frag_b(&Vs[0][0], 72, nt * 16, kc * 32, lane);
          ctx[mt][nt] = wmma_f16(ap, bv, ctx[mt][nt]);
        }
      }
    }
  }

  // normalize and store merged-head ctx [B,S,E] f16
#pragma unroll
  for (int mt = 0; mt < 2; ++mt)
#pragma unroll
    for (int nt = 0; nt < 4; ++nt) {
      const int dk = nt * 16 + (lane & 15);
#pragma unroll
      for (int j = 0; j < 8; ++j) {
        const int row = q0 + wrow0 + mt * 16 + j + sel * 8;
        const size_t idx =
            ((size_t)b * S_DIM + row) * E_DIM + h * DK_DIM + dk;
        Ctx[idx] = (_Float16)(ctx[mt][nt][j] / lst[mt][j]);
      }
    }
}

// ---------------------------------------------------------------------------
// Kernel 3: output projection. out = ctx(f16) * WoT + bo, fp32 out.
// Both tiles are f16 -> both staged via async-LDS.
// ---------------------------------------------------------------------------
__global__ __launch_bounds__(256) void k_out_gemm(
    const _Float16* __restrict__ Xh, const _Float16* __restrict__ WT,
    const float* __restrict__ bias, float* __restrict__ Yf) {
  __shared__ alignas(16) _Float16 As[128][40];
  __shared__ alignas(16) _Float16 Bs[128][40];

  const int t    = threadIdx.x;
  const int lane = t & 31;
  const int wid  = t >> 5;
  const int wm   = wid >> 1;
  const int wn   = wid & 1;
  const int m0   = blockIdx.x * 128;
  const int n0   = blockIdx.y * 128;
  const int sel  = (lane >> 4) & 1;

  v8f acc[2][4];
  v8f z = {};
#pragma unroll
  for (int i = 0; i < 2; ++i)
#pragma unroll
    for (int j = 0; j < 4; ++j) acc[i][j] = z;

  const int row2 = t >> 1, hr = (t & 1) * 16;

  for (int k0 = 0; k0 < E_DIM; k0 += 32) {
    __syncthreads();
    const _Float16* xa = Xh + (size_t)(m0 + row2) * E_DIM + k0 + hr;
#pragma unroll
    for (int c = 0; c < 2; ++c) cp16(&As[row2][hr + c * 8], xa + c * 8);
    const _Float16* xw = WT + (size_t)(n0 + row2) * E_DIM + k0 + hr;
#pragma unroll
    for (int c = 0; c < 2; ++c) cp16(&Bs[row2][hr + c * 8], xw + c * 8);
    cp16_wait();
    __syncthreads();

    v16h af[2];
#pragma unroll
    for (int mt = 0; mt < 2; ++mt)
      af[mt] = ld_frag_a(&As[0][0], 40, wm * 32 + mt * 16, 0, lane);
#pragma unroll
    for (int nt = 0; nt < 4; ++nt) {
      v16h bf = ld_frag_b(&Bs[0][0], 40, wn * 64 + nt * 16, 0, lane);
#pragma unroll
      for (int mt = 0; mt < 2; ++mt)
        acc[mt][nt] = wmma_f16(af[mt], bf, acc[mt][nt]);
    }
  }

#pragma unroll
  for (int mt = 0; mt < 2; ++mt)
#pragma unroll
    for (int nt = 0; nt < 4; ++nt) {
      const int col = n0 + wn * 64 + nt * 16 + (lane & 15);
      const float bb = bias[col];
#pragma unroll
      for (int j = 0; j < 8; ++j) {
        const int row = m0 + wm * 32 + mt * 16 + j + sel * 8;
        Yf[(size_t)row * E_DIM + col] = acc[mt][nt][j] + bb;
      }
    }
}

// ---------------------------------------------------------------------------
extern "C" void kernel_launch(void* const* d_in, const int* in_sizes, int n_in,
                              void* d_out, int out_size, void* d_ws,
                              size_t ws_size, hipStream_t stream) {
  (void)in_sizes; (void)n_in; (void)out_size; (void)ws_size;
  const float* query = (const float*)d_in[0];
  const float* key   = (const float*)d_in[1];
  const float* value = (const float*)d_in[2];
  // d_in[3] = attn_mask (causal, implemented analytically in-kernel)
  const float* Wq = (const float*)d_in[4];
  const float* bq = (const float*)d_in[5];
  const float* Wk = (const float*)d_in[6];
  const float* bk = (const float*)d_in[7];
  const float* Wv = (const float*)d_in[8];
  const float* bv = (const float*)d_in[9];
  const float* Wo = (const float*)d_in[10];
  const float* bo = (const float*)d_in[11];
  float* out = (float*)d_out;

  const size_t NELEM = (size_t)B_DIM * H_DIM * S_DIM * DK_DIM;  // 4.19M
  const size_t WELEM = (size_t)E_DIM * E_DIM;                   // 1.05M
  _Float16* Qp  = (_Float16*)d_ws;
  _Float16* Kp  = Qp + NELEM;
  _Float16* Vp  = Kp + NELEM;
  _Float16* Cx  = Vp + NELEM;
  _Float16* WqT = Cx + NELEM;
  _Float16* WkT = WqT + WELEM;
  _Float16* WvT = WkT + WELEM;
  _Float16* WoT = WvT + WELEM;

  const dim3 gW(E_DIM / 32, E_DIM / 32), bW(256);
  k_wcvt<<<gW, bW, 0, stream>>>(Wq, WqT);
  k_wcvt<<<gW, bW, 0, stream>>>(Wk, WkT);
  k_wcvt<<<gW, bW, 0, stream>>>(Wv, WvT);
  k_wcvt<<<gW, bW, 0, stream>>>(Wo, WoT);

  const dim3 gG(M_TOTAL / 128, E_DIM / 128), bG(256);
  const float qscale = 0.125f;  // 1/sqrt(DK)
  k_proj_gemm<<<gG, bG, 0, stream>>>(query, WqT, bq, Qp, qscale);
  k_proj_gemm<<<gG, bG, 0, stream>>>(key,   WkT, bk, Kp, 1.0f);
  k_proj_gemm<<<gG, bG, 0, stream>>>(value, WvT, bv, Vp, 1.0f);

  const dim3 gA(S_DIM / 128, B_DIM * H_DIM), bA(128);
  k_attn<<<gA, bA, 0, stream>>>(Qp, Kp, Vp, Cx);

  k_out_gemm<<<gG, bG, 0, stream>>>(Cx, WoT, bo, out);
}
